// SadgaLayer_42649025249985
// MI455X (gfx1250) — matrix operand order repaired and checked by reading, not compiled, not run on hardware
//
#include <hip/hip_runtime.h>
#include <hip/hip_bf16.h>
#include <math.h>

// Problem constants (reference: D=256, Q=256, K=512)
#define DD   256
#define QQ   256
#define KK   512
#define SCALE 0.0625f   // 1/sqrt(256)

typedef __attribute__((ext_vector_type(16))) __bf16 v16bf;
typedef __attribute__((ext_vector_type(8)))  float  v8f;

__device__ __forceinline__ float sigmoidf_(float x){ return 1.0f/(1.0f+__expf(-x)); }

__device__ __forceinline__ v8f wmma_bf16(v16bf a, v16bf b, v8f c){
  // 8 args: (neg_a, A, neg_b, B, c_mod, C, reuse_a, reuse_b)
  return __builtin_amdgcn_wmma_f32_16x16x32_bf16(false, a, false, b, (short)0, c, false, false);
}

// A fragment: 16x32 bf16 (M x K slab), row-major f32 source, contiguous layout:
// lane<16: M=lane, K = e
// lane>=16: M=lane-16, K = 16+e
__device__ __forceinline__ v16bf load_a_frag(const float* A, int lda, int m0, int kk, int lane){
  v16bf a;
  int m  = m0 + (lane & 15);
  int hi = (lane >= 16) ? 16 : 0;
  const float* row = A + (size_t)m * lda + kk;
#pragma unroll
  for (int e = 0; e < 16; ++e){
    int k = hi + e;
    a[e] = (__bf16)row[k];
  }
  return a;
}

// B fragment: 32x16 bf16 (K x N slab), row-major f32 source:
// lane<16: N=lane, K = e ; lane>=16: N=lane-16, K = 16+e
__device__ __forceinline__ v16bf load_b_frag(const float* B, int ldb, int kk, int n0, int lane){
  v16bf b;
  int n  = n0 + (lane & 15);
  int hi = (lane >= 16) ? 16 : 0;
#pragma unroll
  for (int e = 0; e < 16; ++e){
    int k = kk + hi + e;
    b[e] = (__bf16)B[(size_t)k * ldb + n];
  }
  return b;
}

// ---------------- Kernel 1: prep (q_glob_enc mean, t, g1, k_s) ----------------
__global__ void prep_kernel(const float* __restrict__ qenc, const float* __restrict__ Wglob,
                            const float* __restrict__ bglob, const float* __restrict__ Wgv0,
                            const float* __restrict__ bgv0, const float* __restrict__ kenc,
                            float* __restrict__ g1, float* __restrict__ ks){
  __shared__ float s_qge[DD], s_t[DD];
  int d = threadIdx.x;
  float s = 0.0f;
  for (int q = 0; q < QQ; ++q) s += qenc[(size_t)q*DD + d];
  s *= (1.0f/(float)QQ);
  s_qge[d] = s;
  __syncthreads();
  float tv = bglob[d], gv = bgv0[d];
  for (int e = 0; e < DD; ++e){
    float x = s_qge[e];
    tv += x * Wglob[(size_t)e*DD + d];
    gv += x * Wgv0[(size_t)e*DD + d];
  }
  s_t[d] = tv;
  g1[d]  = gv;
  __syncthreads();
#pragma unroll
  for (int rep = 0; rep < 2; ++rep){
    int k = d + rep*DD;
    float dot = 0.0f;
    for (int e = 0; e < DD; ++e) dot += s_t[e] * kenc[(size_t)k*DD + e];
    ks[k] = sigmoidf_(dot * SCALE);
  }
}

// ------------- Kernel 2: k_enc2 = (1-ks)*(k_enc@Wgv1+b) + ks*g1 (WMMA) -------------
__global__ void kenc2_kernel(const float* __restrict__ kenc, const float* __restrict__ Wgv1,
                             const float* __restrict__ bgv1, const float* __restrict__ ks,
                             const float* __restrict__ g1, float* __restrict__ K2){
  int lane = threadIdx.x & 31, wave = threadIdx.x >> 5;
  int tile = blockIdx.x * 8 + wave;           // M=512,N=256 -> 32x16=512 tiles, 64 blocks
  int tm = tile >> 4, tn = tile & 15;
  int m0 = tm << 4, n0 = tn << 4;
  v8f acc = {};
  for (int kk = 0; kk < DD; kk += 32){
    v16bf a = load_a_frag(kenc, DD, m0, kk, lane);
    v16bf b = load_b_frag(Wgv1, DD, kk, n0, lane);
    acc = wmma_bf16(a, b, acc);
  }
  int col = n0 + (lane & 15);
  int r0  = m0 + ((lane >= 16) ? 8 : 0);
  float g = g1[col], bb = bgv1[col];
#pragma unroll
  for (int r = 0; r < 8; ++r){
    int row = r0 + r;
    float sgl = ks[row];
    K2[(size_t)row*DD + col] = (1.0f - sgl)*(acc[r] + bb) + sgl*g;
  }
}

// ------------- Kernel 3: generic C = A@B + bias (WMMA) — q_glob / q_loc -------------
__global__ void gemm_bias_kernel(const float* __restrict__ A, const float* __restrict__ B,
                                 const float* __restrict__ bias, float* __restrict__ C,
                                 int M, int N, int Kd){
  int lane = threadIdx.x & 31, wave = threadIdx.x >> 5;
  int tilesN = N >> 4;
  int tile = blockIdx.x * 8 + wave;
  int tm = tile / tilesN, tn = tile % tilesN;
  int m0 = tm << 4, n0 = tn << 4;
  if (m0 >= M) return;
  v8f acc = {};
  for (int kk = 0; kk < Kd; kk += 32){
    v16bf a = load_a_frag(A, Kd, m0, kk, lane);
    v16bf b = load_b_frag(B, N, kk, n0, lane);
    acc = wmma_bf16(a, b, acc);
  }
  int col = n0 + (lane & 15);
  int r0  = m0 + ((lane >= 16) ? 8 : 0);
  float bb = bias[col];
#pragma unroll
  for (int r = 0; r < 8; ++r) C[(size_t)(r0 + r)*N + col] = acc[r] + bb;
}

// ------- Kernel 4: stream rel_k; fused logits E=exp(tanh(.)) + glob_attn softmax -------
// block = one q (256 threads, 8 waves); wave handles k = wave, wave+8, ...
__global__ void logits_kernel(const float* __restrict__ qg, const float* __restrict__ ql,
                              const float* __restrict__ K2, const float* __restrict__ relk,
                              float* __restrict__ E, float* __restrict__ GA){
  __shared__ float s_qg[DD], s_ql[DD], s_e1[KK], s_e2[KK], s_red[8];
  int q = blockIdx.x, tid = threadIdx.x;
  s_qg[tid] = qg[(size_t)q*DD + tid];
  s_ql[tid] = ql[(size_t)q*DD + tid];
  __syncthreads();
  int lane = tid & 31, wave = tid >> 5;
  const float* relq = relk + (size_t)q * KK * DD;
  float gl[8], ll[8];
#pragma unroll
  for (int t = 0; t < 8; ++t){ gl[t] = s_qg[lane*8 + t]; ll[t] = s_ql[lane*8 + t]; }
  for (int k = wave; k < KK; k += 8){
    const float4* r4 = (const float4*)(relq + (size_t)k*DD) + lane*2;
    const float4* c4 = (const float4*)(K2   + (size_t)k*DD) + lane*2;
    __builtin_prefetch(relq + (size_t)(k+8)*DD + lane*8, 0, 0);  // speculative stream-ahead
    float4 ra = r4[0], rb = r4[1], ca = c4[0], cb = c4[1];
    float v[8] = { ra.x+ca.x, ra.y+ca.y, ra.z+ca.z, ra.w+ca.w,
                   rb.x+cb.x, rb.y+cb.y, rb.z+cb.z, rb.w+cb.w };
    float s1 = 0.0f, s2 = 0.0f;
#pragma unroll
    for (int t = 0; t < 8; ++t){ s1 += v[t]*gl[t]; s2 += v[t]*ll[t]; }
#pragma unroll
    for (int off = 16; off > 0; off >>= 1){
      s1 += __shfl_xor(s1, off, 32);
      s2 += __shfl_xor(s2, off, 32);
    }
    if (lane == 0){
      s_e1[k] = __expf(tanhf(s1 * SCALE));   // glob branch (unnormalized softmax numer)
      s_e2[k] = __expf(tanhf(s2 * SCALE));   // E[q,k] = exp(loc_attn)
    }
  }
  __syncthreads();
  float p = s_e1[tid] + s_e1[tid + 256];
#pragma unroll
  for (int off = 16; off > 0; off >>= 1) p += __shfl_xor(p, off, 32);
  if (lane == 0) s_red[wave] = p;
  __syncthreads();
  float tot = 0.0f;
#pragma unroll
  for (int w = 0; w < 8; ++w) tot += s_red[w];
  float inv = 1.0f / tot;
  E [(size_t)q*KK + tid]       = s_e2[tid];
  E [(size_t)q*KK + 256 + tid] = s_e2[tid + 256];
  GA[(size_t)q*KK + tid]       = s_e1[tid] * inv;
  GA[(size_t)q*KK + 256 + tid] = s_e1[tid + 256] * inv;
}

// ------------- Kernel 5: u = K2@Wg1, c2 = K2@Wg2 (wave per row) -------------
__global__ void uc2_kernel(const float* __restrict__ K2, const float* __restrict__ Wlocgate,
                           float* __restrict__ u, float* __restrict__ c2){
  int lane = threadIdx.x & 31, wave = threadIdx.x >> 5;
  int k = blockIdx.x * 8 + wave;
  const float* row = K2 + (size_t)k*DD + lane*8;
  float su = 0.0f, sc = 0.0f;
#pragma unroll
  for (int t = 0; t < 8; ++t){
    float v = row[t];
    su += v * Wlocgate[lane*8 + t];
    sc += v * Wlocgate[DD + lane*8 + t];
  }
#pragma unroll
  for (int off = 16; off > 0; off >>= 1){
    su += __shfl_xor(su, off, 32);
    sc += __shfl_xor(sc, off, 32);
  }
  if (lane == 0){ u[k] = su; c2[k] = sc; }
}

// ------- Kernel 6: Z = E@M^T, S = (E*u)@M^T (WMMA, shared B fragment) -------
__global__ void zs_kernel(const float* __restrict__ E, const float* __restrict__ u,
                          const int* __restrict__ adj, float* __restrict__ Zb,
                          float* __restrict__ Sb){
  int lane = threadIdx.x & 31, wave = threadIdx.x >> 5;
  int tile = blockIdx.x * 8 + wave;           // M=256(q), N=512(i) -> 16x32=512 tiles
  int tm = tile >> 5, tn = tile & 31;
  int m0 = tm << 4, n0 = tn << 4;
  int mrow = m0 + (lane & 15);
  int hiA  = (lane >= 16) ? 16 : 0;
  int bn   = n0 + (lane & 15);
  int hiB  = (lane >= 16) ? 16 : 0;
  v8f az = {}, as = {};
  for (int kk = 0; kk < KK; kk += 32){
    v16bf ae, af, b;
#pragma unroll
    for (int e = 0; e < 16; ++e){
      int j = kk + hiA + e;
      float ev = E[(size_t)mrow*KK + j];
      ae[e] = (__bf16)ev;
      af[e] = (__bf16)(ev * u[j]);
      int jb = kk + hiB + e;                              // B[j,i] = M[i,j] (transposed read)
      b[e] = (__bf16)((adj[(size_t)bn*KK + jb] != 0) ? 1.0f : 0.0f);
    }
    az = wmma_bf16(ae, b, az);
    as = wmma_bf16(af, b, as);
  }
  int r0 = m0 + ((lane >= 16) ? 8 : 0);
#pragma unroll
  for (int r = 0; r < 8; ++r){
    Zb[(size_t)(r0 + r)*KK + bn] = az[r];
    Sb[(size_t)(r0 + r)*KK + bn] = as[r];
  }
}

// ------------- Kernel 7: loc_gate, AZ = ga*lg/Z, CR = ga*(1-lg) -------------
__global__ void locgate_kernel(const float* __restrict__ Zb, const float* __restrict__ Sb,
                               const float* __restrict__ c2, const float* __restrict__ GA,
                               const float* __restrict__ bl, float* __restrict__ AZ,
                               float* __restrict__ CR){
  int idx = blockIdx.x * 256 + threadIdx.x;   // Q*K = 131072
  int i = idx & (KK - 1);
  float z  = Zb[idx];
  float zs = (z > 1e-30f) ? z : 1e-30f;       // mask rows are never empty in practice
  float lg = sigmoidf_(Sb[idx]/zs + c2[i] + bl[0]);
  float ga = GA[idx];
  AZ[idx] = ga * lg / zs;
  CR[idx] = ga * (1.0f - lg);
}

// ------- Kernel 8: B2 = AZ@M (WMMA); coeff = B2*E + CR in epilogue -------
__global__ void b2coeff_kernel(const float* __restrict__ AZ, const int* __restrict__ adj,
                               const float* __restrict__ E, const float* __restrict__ CR,
                               float* __restrict__ coeff){
  int lane = threadIdx.x & 31, wave = threadIdx.x >> 5;
  int tile = blockIdx.x * 8 + wave;           // M=256(q), N=512(j)
  int tm = tile >> 5, tn = tile & 31;
  int m0 = tm << 4, n0 = tn << 4;
  int bn = n0 + (lane & 15);
  int hiB = (lane >= 16) ? 16 : 0;
  v8f acc = {};
  for (int kk = 0; kk < KK; kk += 32){
    v16bf a = load_a_frag(AZ, KK, m0, kk, lane);
    v16bf b;
#pragma unroll
    for (int e = 0; e < 16; ++e){
      int ib = kk + hiB + e;                               // B[i,j] = M[i,j] direct
      b[e] = (__bf16)((adj[(size_t)ib*KK + bn] != 0) ? 1.0f : 0.0f);
    }
    acc = wmma_bf16(a, b, acc);
  }
  int r0 = m0 + ((lane >= 16) ? 8 : 0);
#pragma unroll
  for (int r = 0; r < 8; ++r){
    size_t idx = (size_t)(r0 + r)*KK + bn;
    coeff[idx] = acc[r]*E[idx] + CR[idx];
  }
}

// ------------- Kernel 9: stream rel_v via GLOBAL_LOAD_ASYNC_TO_LDS (double-buffered) -------------
// KRM[q,d] = sum_i GA[q,i]*rel_v[q,i,d]. Each thread owns a private 16B LDS slot:
// it async-copies rel_v[q, i+isub, d4*4..+3] into its slot, then reads it back, so only
// the per-wave s_wait_asynccnt is required (async loads complete in order; waiting for
// count<=1 after issuing the next tile guarantees the current tile has landed).
__global__ void krm_kernel(const float* __restrict__ GA, const float* __restrict__ relv,
                           float* __restrict__ KRM){
  __shared__ float  s_ga[KK];
  __shared__ float  s_buf[2][1024];   // double buffer: 4 rows x 256 floats each
  __shared__ float4 s_acc[4][64];
  int q = blockIdx.x, tid = threadIdx.x;
  s_ga[tid]       = GA[(size_t)q*KK + tid];
  s_ga[tid + 256] = GA[(size_t)q*KK + 256 + tid];
  int isub = tid >> 6, d4 = tid & 63;
  const float* gsrc = relv + (size_t)q*KK*DD + (size_t)isub*DD + (size_t)d4*4;
  unsigned lds0 = (unsigned)(uintptr_t)&s_buf[0][tid*4];
  unsigned lds1 = (unsigned)(uintptr_t)&s_buf[1][tid*4];
  // kick off first tile (ASYNCcnt = 1)
  asm volatile("global_load_async_to_lds_b128 %0, %1, off"
               :: "v"(lds0), "v"(gsrc) : "memory");
  __syncthreads();   // s_ga ready
  float ax = 0.f, ay = 0.f, az = 0.f, aw = 0.f;
  for (int i = 0; i < KK; i += 4){
    unsigned curbuf = (i >> 2) & 1;
    if (i + 4 < KK){
      asm volatile("global_load_async_to_lds_b128 %0, %1, off"
                   :: "v"(curbuf ? lds0 : lds1), "v"(gsrc + (size_t)(i + 4)*DD) : "memory");
      asm volatile("s_wait_asynccnt 0x1" ::: "memory");   // current tile done, next in flight
    } else {
      asm volatile("s_wait_asynccnt 0x0" ::: "memory");   // last tile: drain
    }
    float4 v = *(const float4*)&s_buf[curbuf][tid*4];
    float w = s_ga[i + isub];
    ax += w*v.x; ay += w*v.y; az += w*v.z; aw += w*v.w;
  }
  s_acc[isub][d4] = make_float4(ax, ay, az, aw);
  __syncthreads();
  if (isub == 0){
    float4 a = s_acc[0][d4], b = s_acc[1][d4], c = s_acc[2][d4], e = s_acc[3][d4];
    float* out = KRM + (size_t)q*DD + d4*4;
    out[0] = a.x + b.x + c.x + e.x;
    out[1] = a.y + b.y + c.y + e.y;
    out[2] = a.z + b.z + c.z + e.z;
    out[3] = a.w + b.w + c.w + e.w;
  }
}

// ------- Kernel 10: q_enc_new = coeff@K2 + KRM (WMMA) -------
__global__ void newq_kernel(const float* __restrict__ coeff, const float* __restrict__ K2,
                            const float* __restrict__ KRM, float* __restrict__ NQ){
  int lane = threadIdx.x & 31, wave = threadIdx.x >> 5;
  int tile = blockIdx.x * 8 + wave;           // M=256, N=256 -> 256 tiles, 32 blocks
  int tm = tile >> 4, tn = tile & 15;
  int m0 = tm << 4, n0 = tn << 4;
  v8f acc = {};
  for (int kk = 0; kk < KK; kk += 32){
    v16bf a = load_a_frag(coeff, KK, m0, kk, lane);
    v16bf b = load_b_frag(K2, DD, kk, n0, lane);
    acc = wmma_bf16(a, b, acc);
  }
  int col = n0 + (lane & 15);
  int r0  = m0 + ((lane >= 16) ? 8 : 0);
#pragma unroll
  for (int r = 0; r < 8; ++r){
    size_t idx = (size_t)(r0 + r)*DD + col;
    NQ[idx] = acc[r] + KRM[idx];
  }
}

// ------------- Kernel 11: final gate + output h -------------
__global__ void final_kernel(const float* __restrict__ NQ, const float* __restrict__ qenc,
                             const float* __restrict__ Wgate, const float* __restrict__ bg,
                             float* __restrict__ h){
  __shared__ float s_red[8];
  int q = blockIdx.x, tid = threadIdx.x;
  float n = NQ  [(size_t)q*DD + tid];
  float e = qenc[(size_t)q*DD + tid];
  float p = n*Wgate[tid] + e*Wgate[DD + tid];
  int lane = tid & 31, wave = tid >> 5;
#pragma unroll
  for (int off = 16; off > 0; off >>= 1) p += __shfl_xor(p, off, 32);
  if (lane == 0) s_red[wave] = p;
  __syncthreads();
  float tot = 0.0f;
#pragma unroll
  for (int w = 0; w < 8; ++w) tot += s_red[w];
  float g = sigmoidf_(tot + bg[0]);
  h[(size_t)q*DD + tid] = g*n + (1.0f - g)*e;
}

extern "C" void kernel_launch(void* const* d_in, const int* in_sizes, int n_in,
                              void* d_out, int out_size, void* d_ws, size_t ws_size,
                              hipStream_t stream){
  (void)in_sizes; (void)n_in; (void)out_size; (void)ws_size;
  const float* q_enc     = (const float*)d_in[0];
  const float* k_enc     = (const float*)d_in[1];
  const float* rel_k     = (const float*)d_in[2];
  const float* rel_v     = (const float*)d_in[3];
  const int*   k_adj     = (const int*)  d_in[4];
  const float* W_glob    = (const float*)d_in[5];
  const float* b_glob    = (const float*)d_in[6];
  const float* W_gv0     = (const float*)d_in[7];
  const float* b_gv0     = (const float*)d_in[8];
  const float* W_gv1     = (const float*)d_in[9];
  const float* b_gv1     = (const float*)d_in[10];
  const float* W_gq      = (const float*)d_in[11];
  const float* b_gq      = (const float*)d_in[12];
  const float* W_lq      = (const float*)d_in[13];
  const float* b_lq      = (const float*)d_in[14];
  const float* W_locgate = (const float*)d_in[15];
  const float* b_locgate = (const float*)d_in[16];
  const float* W_gate    = (const float*)d_in[17];
  const float* b_gate    = (const float*)d_in[18];
  float* out = (float*)d_out;

  float* ws = (float*)d_ws;      // ~5.25 MB of f32 scratch used
  float* g1    = ws;             // 256
  float* ks    = ws + 256;       // 512
  float* K2    = ws + 768;       // 512*256
  float* qg    = K2   + KK*DD;   // 256*256
  float* ql    = qg   + QQ*DD;   // 256*256
  float* E     = ql   + QQ*DD;   // 256*512
  float* GA    = E    + QQ*KK;   // 256*512
  float* u     = GA   + QQ*KK;   // 512
  float* c2    = u    + KK;      // 512
  float* Zb    = c2   + KK;      // 256*512
  float* Sb    = Zb   + QQ*KK;   // 256*512
  float* AZ    = Sb   + QQ*KK;   // 256*512
  float* CR    = AZ   + QQ*KK;   // 256*512
  float* coeff = CR   + QQ*KK;   // 256*512
  float* KRM   = coeff+ QQ*KK;   // 256*256
  float* NQ    = KRM  + QQ*DD;   // 256*256

  prep_kernel   <<<1,   256, 0, stream>>>(q_enc, W_glob, b_glob, W_gv0, b_gv0, k_enc, g1, ks);
  kenc2_kernel  <<<64,  256, 0, stream>>>(k_enc, W_gv1, b_gv1, ks, g1, K2);
  gemm_bias_kernel<<<32,256, 0, stream>>>(q_enc, W_gq, b_gq, qg, QQ, DD, DD);
  gemm_bias_kernel<<<32,256, 0, stream>>>(q_enc, W_lq, b_lq, ql, QQ, DD, DD);
  logits_kernel <<<QQ,  256, 0, stream>>>(qg, ql, K2, rel_k, E, GA);
  uc2_kernel    <<<64,  256, 0, stream>>>(K2, W_locgate, u, c2);
  zs_kernel     <<<64,  256, 0, stream>>>(E, u, k_adj, Zb, Sb);
  locgate_kernel<<<512, 256, 0, stream>>>(Zb, Sb, c2, GA, b_locgate, AZ, CR);
  b2coeff_kernel<<<64,  256, 0, stream>>>(AZ, k_adj, E, CR, coeff);
  krm_kernel    <<<QQ,  256, 0, stream>>>(GA, rel_v, KRM);
  newq_kernel   <<<32,  256, 0, stream>>>(coeff, K2, KRM, NQ);
  final_kernel  <<<QQ,  256, 0, stream>>>(NQ, q_enc, W_gate, b_gate, out);
}
